// TemporalGAT_51470888075728
// MI455X (gfx1250) — matrix operand order repaired and checked by reading, not compiled, not run on hardware
//
#include <hip/hip_runtime.h>
#include <math.h>

typedef __attribute__((ext_vector_type(16))) _Float16 v16h;
typedef __attribute__((ext_vector_type(8)))  float    v8f;

#define ACT_NONE 0
#define ACT_ELU  1
#define ACT_RELU 2

// ---- order-preserving float<->uint encoding for atomicMax-based segment max ----
__device__ __forceinline__ unsigned fenc(float f) {
    unsigned u = __float_as_uint(f);
    return (u & 0x80000000u) ? ~u : (u | 0x80000000u);
}
__device__ __forceinline__ float fdec(unsigned u) {
    return __uint_as_float((u & 0x80000000u) ? (u & 0x7fffffffu) : ~u);
}
#define ENC_NEG_INF 0x007fffffu   // fenc(-inf)

// ---- f32 -> f16 conversion (weights once per launch, activations per step) ----
__global__ void cvt_f16_k(const float* __restrict__ src, _Float16* __restrict__ dst,
                          long long n)
{
    long long i = (long long)blockIdx.x * blockDim.x + threadIdx.x;
    if (i < n) dst[i] = (_Float16)src[i];
}

// =======================================================================
// GEMM: Y[M,Kout] = act(X[M,Kin] @ W[Kout,Kin]^T + bias), X/W in f16.
// Grid: (ceil(ntiles/8), mtiles). One wave32 per 16x16 tile.
// A tile (16 rows x Kin) staged in LDS once per block, shared by 8 waves;
// fragments read as ds_load_b128 / global b128 loads; f32 accumulate.
// Requires M % 16 == 0, Kin % 32 == 0 (<=256), Kout % 16 == 0.
// =======================================================================
template<int ACT, bool BIAS>
__global__ void gemm_wmma_k(const _Float16* __restrict__ X, const _Float16* __restrict__ W,
                            const float* __restrict__ bias, float* __restrict__ Y,
                            int M, int Kin, int Kout)
{
    __shared__ __align__(16) _Float16 lA[16 * 256];   // up to 8KB of 320KB/WGP

    const int mt   = blockIdx.y;
    const int lane = threadIdx.x & 31;
    const int ntiles = Kout >> 4;
    const int nt   = blockIdx.x * 8 + (threadIdx.x >> 5);

    // ---- cooperative stage of A tile: 16 rows x Kin halves ----
    {
        const uint4* gsrc = (const uint4*)(X + (long long)mt * 16 * Kin);
        uint4* ldst = (uint4*)lA;
        int chunks = (16 * Kin) >> 3;                 // 8 halves per uint4
        for (int i = threadIdx.x; i < chunks; i += blockDim.x) ldst[i] = gsrc[i];
    }
    __syncthreads();

    if (nt >= ntiles) return;                         // wave-uniform: EXEC all-1s below
    const int l = lane & 15, hi = lane >> 4;
    const _Float16* wrow = W + (long long)(nt * 16 + l) * Kin;   // B col N = lane&15

    v8f acc = {};
    for (int kb = 0; kb < Kin; kb += 32) {
        if (kb + 32 < Kin) __builtin_prefetch(wrow + kb + 32, 0, 1);  // global_prefetch_b8
        // A fragment (16-bit A 16x32): K = kb + 16*hi + e, 16 contiguous halves
        union { v16h v; uint4 q[2]; } ua;
        ua.q[0] = *(const uint4*)&lA[l * Kin + kb + 16 * hi];
        ua.q[1] = *(const uint4*)&lA[l * Kin + kb + 16 * hi + 8];
        // B fragment (16-bit B 32x16, same k order as A): K = kb + 16*hi + e, 16 contiguous halves
        union { v16h v; uint4 q[2]; } ub;
        ub.q[0] = *(const uint4*)&wrow[kb + 16 * hi];
        ub.q[1] = *(const uint4*)&wrow[kb + 16 * hi + 8];
        acc = __builtin_amdgcn_wmma_f32_16x16x32_f16(false, ua.v, false, ub.v,
                                                     (short)0, acc, false, false);
    }
#pragma unroll
    for (int r = 0; r < 8; ++r) {
        int orow = mt * 16 + r + 8 * hi;              // C/D: lanes16-31 hold M=r+8
        int ocol = nt * 16 + l;
        float v = acc[r];
        if (BIAS) v += bias[ocol];
        if (ACT == ACT_ELU)  v = (v > 0.f) ? v : (__expf(v) - 1.f);
        if (ACT == ACT_RELU) v = fmaxf(v, 0.f);
        Y[(long long)orow * Kout + ocol] = v;
    }
}

// ---- es[n,h] = <h[n,h,:], a_src[h,:]> ; ed likewise -------------------
__global__ void attn_scores_k(const float* __restrict__ H, const float* __restrict__ asrc,
                              const float* __restrict__ adst, float* __restrict__ es,
                              float* __restrict__ ed, int n, int heads, int C)
{
    int i = blockIdx.x * blockDim.x + threadIdx.x;
    if (i >= n * heads) return;
    int node = i / heads, h = i - node * heads;
    const float* hp = H + (long long)node * heads * C + h * C;
    float s = 0.f, d = 0.f;
#pragma unroll 4
    for (int c = 0; c < C; ++c) { float v = hp[c]; s += v * asrc[h * C + c]; d += v * adst[h * C + c]; }
    es[i] = s; ed[i] = d;
}

__global__ void fill_f32_k(float* p, float v, long long n) {
    long long i = (long long)blockIdx.x * blockDim.x + threadIdx.x;
    if (i < n) p[i] = v;
}
__global__ void fill_u32_k(unsigned* p, unsigned v, long long n) {
    long long i = (long long)blockIdx.x * blockDim.x + threadIdx.x;
    if (i < n) p[i] = v;
}

// ---- pass 1: per-dst segment max of leaky_relu(es[src]+ed[dst]) -------
__global__ void edge_max_k(const int* __restrict__ ei, int E, int N,
                           const float* __restrict__ es, const float* __restrict__ ed,
                           unsigned* __restrict__ m, int heads)
{
    int e = blockIdx.x * blockDim.x + threadIdx.x;
    int Etot = E + N;
    if (e >= Etot) return;
    int s = (e < E) ? ei[e]                : (e - E);   // self-loops appended
    int d = (e < E) ? ei[(long long)E + e] : (e - E);
    for (int h = 0; h < heads; ++h) {
        float v = es[s * heads + h] + ed[d * heads + h];
        v = (v > 0.f) ? v : 0.2f * v;                   // leaky_relu(0.2)
        atomicMax(&m[d * heads + h], fenc(v));
    }
}

// ---- pass 2: ex = exp(e - max[dst]); den[dst] += ex; stash ex ---------
__global__ void edge_exp_k(const int* __restrict__ ei, int E, int N,
                           const float* __restrict__ es, const float* __restrict__ ed,
                           const unsigned* __restrict__ m, float* __restrict__ den,
                           float* __restrict__ alpha, int heads)
{
    int e = blockIdx.x * blockDim.x + threadIdx.x;
    int Etot = E + N;
    if (e >= Etot) return;
    int s = (e < E) ? ei[e]                : (e - E);
    int d = (e < E) ? ei[(long long)E + e] : (e - E);
    for (int h = 0; h < heads; ++h) {
        float v = es[s * heads + h] + ed[d * heads + h];
        v = (v > 0.f) ? v : 0.2f * v;
        float ex = __expf(v - fdec(m[d * heads + h]));
        alpha[(long long)e * heads + h] = ex;
        atomicAdd(&den[d * heads + h], ex);
    }
}

// ---- pass 3: out[dst] += h[src] * alpha/den ; one thread per (edge, c) --
__global__ void edge_scatter_k(const int* __restrict__ ei, int E, int N,
                               const float* __restrict__ H, const float* __restrict__ alpha,
                               const float* __restrict__ den, float* __restrict__ out,
                               int heads, int C)
{
    long long idx = (long long)blockIdx.x * blockDim.x + threadIdx.x;
    long long Etot = (long long)E + N;
    if (idx >= Etot * C) return;
    int e = (int)(idx / C);
    int c = (int)(idx - (long long)e * C);
    int s = (e < E) ? ei[e]                : (e - E);
    int d = (e < E) ? ei[(long long)E + e] : (e - E);
    for (int h = 0; h < heads; ++h) {
        float coef = alpha[(long long)e * heads + h] / den[d * heads + h];
        float v = H[(long long)s * heads * C + h * C + c] * coef;
        atomicAdd(&out[(long long)d * heads * C + h * C + c], v);
    }
}

// ---- bias + optional ELU, in place ------------------------------------
template<int ACT>
__global__ void bias_act_k(float* __restrict__ Y, const float* __restrict__ bias,
                           long long M, int K)
{
    long long i = (long long)blockIdx.x * blockDim.x + threadIdx.x;
    if (i >= M * K) return;
    float v = Y[i] + bias[(int)(i % K)];
    if (ACT == ACT_ELU)  v = (v > 0.f) ? v : (__expf(v) - 1.f);
    if (ACT == ACT_RELU) v = fmaxf(v, 0.f);
    Y[i] = v;
}

// ---- fused GRU gates: PyTorch [r,z,n] order ---------------------------
__global__ void gru_gate_k(const float* __restrict__ gi, const float* __restrict__ gh,
                           float* __restrict__ hid, int n, int Hd)
{
    int i = blockIdx.x * blockDim.x + threadIdx.x;
    if (i >= n * Hd) return;
    int node = i / Hd, j = i - node * Hd;
    const float* gip = gi + (long long)node * 3 * Hd;
    const float* ghp = gh + (long long)node * 3 * Hd;
    float r  = 1.f / (1.f + __expf(-(gip[j]          + ghp[j])));
    float z  = 1.f / (1.f + __expf(-(gip[Hd + j]     + ghp[Hd + j])));
    float nn = tanhf(gip[2 * Hd + j] + r * ghp[2 * Hd + j]);
    hid[i] = (1.f - z) * nn + z * hid[i];
}

// ---- final linear: logits[n] = <tmp[n,:], Wm2> + bm2 ------------------
__global__ void mlp_out_k(const float* __restrict__ tmp, const float* __restrict__ Wm2,
                          const float* __restrict__ bm2, float* __restrict__ out, int n)
{
    int i = blockIdx.x * blockDim.x + threadIdx.x;
    if (i >= n) return;
    float s = bm2[0];
#pragma unroll 4
    for (int c = 0; c < 64; ++c) s += tmp[(long long)i * 64 + c] * Wm2[c];
    out[i] = s;
}

// =======================================================================
static inline dim3 gemm_grid(int M, int Kout) {
    return dim3((unsigned)(((Kout >> 4) + 7) / 8), (unsigned)(M >> 4));
}
static inline unsigned cdiv_ll(long long n, int b) { return (unsigned)((n + b - 1) / b); }

extern "C" void kernel_launch(void* const* d_in, const int* in_sizes, int n_in,
                              void* d_out, int out_size, void* d_ws, size_t ws_size,
                              hipStream_t stream)
{
    (void)n_in; (void)out_size; (void)ws_size;
    const float* x_seq = (const float*)d_in[0];
    const int*   ei    = (const int*)d_in[1];
    const float* W1    = (const float*)d_in[2];
    const float* a1s   = (const float*)d_in[3];
    const float* a1d   = (const float*)d_in[4];
    const float* b1    = (const float*)d_in[5];
    const float* W2    = (const float*)d_in[6];
    const float* a2s   = (const float*)d_in[7];
    const float* a2d   = (const float*)d_in[8];
    const float* b2    = (const float*)d_in[9];
    const float* Wih   = (const float*)d_in[10];
    const float* Whh   = (const float*)d_in[11];
    const float* bih   = (const float*)d_in[12];
    const float* bhh   = (const float*)d_in[13];
    const float* Wm1   = (const float*)d_in[14];
    const float* bm1   = (const float*)d_in[15];
    const float* Wm2   = (const float*)d_in[16];
    const float* bm2   = (const float*)d_in[17];

    const int T = 8, IN = 64, C = 64, HEADS = 4;
    const int E = in_sizes[1] / 2;
    const int N = in_sizes[0] / (T * IN);        // 20000, divisible by 16
    const long long Etot = (long long)E + N;

    // ---- workspace carve-up: f32 region, then f16 region ----
    float* ws = (float*)d_ws;
    size_t off = 0;
    float*    f_h    = ws + off; off += (size_t)N * 256;   // h1 (later reused for h2)
    float*    f_agg  = ws + off; off += (size_t)N * 256;   // layer-1 aggregate
    float*    f_out2 = ws + off; off += (size_t)N * 64;    // layer-2 aggregate
    float*    f_es   = ws + off; off += (size_t)N * 4;
    float*    f_ed   = ws + off; off += (size_t)N * 4;
    unsigned* f_m    = (unsigned*)(ws + off); off += (size_t)N * 4;
    float*    f_den  = ws + off; off += (size_t)N * 4;
    float*    f_alp  = ws + off; off += (size_t)Etot * 4;
    float*    f_hid  = ws + off; off += (size_t)N * 64;
    float*    f_gi   = ws + off; off += (size_t)N * 192;
    float*    f_gh   = ws + off; off += (size_t)N * 192;
    float*    f_tmp  = ws + off; off += (size_t)N * 64;
    _Float16* hp = (_Float16*)(ws + off);
    size_t ho = 0;
    _Float16* h_A   = hp + ho; ho += (size_t)N * 256;      // current GEMM A operand
    _Float16* h_B   = hp + ho; ho += (size_t)N * 64;       // second GRU operand
    _Float16* h_W1  = hp + ho; ho += 256 * 64;
    _Float16* h_W2  = hp + ho; ho += 64 * 256;
    _Float16* h_Wih = hp + ho; ho += 192 * 64;
    _Float16* h_Whh = hp + ho; ho += 192 * 64;
    _Float16* h_Wm1 = hp + ho; ho += 64 * 64;

    const int B = 256;
    // weights -> f16 once per launch (deterministic: redone every call)
    cvt_f16_k<<<cdiv_ll(256 * 64, B), B, 0, stream>>>(W1,  h_W1,  256 * 64);
    cvt_f16_k<<<cdiv_ll(64 * 256, B), B, 0, stream>>>(W2,  h_W2,  64 * 256);
    cvt_f16_k<<<cdiv_ll(192 * 64, B), B, 0, stream>>>(Wih, h_Wih, 192 * 64);
    cvt_f16_k<<<cdiv_ll(192 * 64, B), B, 0, stream>>>(Whh, h_Whh, 192 * 64);
    cvt_f16_k<<<cdiv_ll(64 * 64,  B), B, 0, stream>>>(Wm1, h_Wm1, 64 * 64);
    // h0 = 0
    fill_f32_k<<<cdiv_ll((long long)N * 64, B), B, 0, stream>>>(f_hid, 0.f, (long long)N * 64);

    for (int t = 0; t < T; ++t) {
        const float* xt = x_seq + (size_t)t * N * IN;

        // ---------- GAT layer 1 ----------
        cvt_f16_k<<<cdiv_ll((long long)N * 64, B), B, 0, stream>>>(xt, h_A, (long long)N * 64);
        gemm_wmma_k<ACT_NONE, false><<<gemm_grid(N, 256), B, 0, stream>>>(
            h_A, h_W1, nullptr, f_h, N, 64, 256);
        attn_scores_k<<<cdiv_ll((long long)N * HEADS, B), B, 0, stream>>>(
            f_h, a1s, a1d, f_es, f_ed, N, HEADS, C);
        fill_u32_k<<<cdiv_ll((long long)N * HEADS, B), B, 0, stream>>>(f_m, ENC_NEG_INF, (long long)N * HEADS);
        fill_f32_k<<<cdiv_ll((long long)N * HEADS, B), B, 0, stream>>>(f_den, 0.f, (long long)N * HEADS);
        fill_f32_k<<<cdiv_ll((long long)N * 256, B), B, 0, stream>>>(f_agg, 0.f, (long long)N * 256);
        edge_max_k<<<cdiv_ll(Etot, B), B, 0, stream>>>(ei, E, N, f_es, f_ed, f_m, HEADS);
        edge_exp_k<<<cdiv_ll(Etot, B), B, 0, stream>>>(ei, E, N, f_es, f_ed, f_m, f_den, f_alp, HEADS);
        edge_scatter_k<<<cdiv_ll(Etot * C, B), B, 0, stream>>>(ei, E, N, f_h, f_alp, f_den, f_agg, HEADS, C);
        bias_act_k<ACT_ELU><<<cdiv_ll((long long)N * 256, B), B, 0, stream>>>(f_agg, b1, N, 256);

        // ---------- GAT layer 2 (1 head) ----------
        cvt_f16_k<<<cdiv_ll((long long)N * 256, B), B, 0, stream>>>(f_agg, h_A, (long long)N * 256);
        gemm_wmma_k<ACT_NONE, false><<<gemm_grid(N, 64), B, 0, stream>>>(
            h_A, h_W2, nullptr, f_h, N, 256, 64);      // f_h now holds h2 [N,64]
        attn_scores_k<<<cdiv_ll((long long)N, B), B, 0, stream>>>(
            f_h, a2s, a2d, f_es, f_ed, N, 1, C);
        fill_u32_k<<<cdiv_ll((long long)N, B), B, 0, stream>>>(f_m, ENC_NEG_INF, (long long)N);
        fill_f32_k<<<cdiv_ll((long long)N, B), B, 0, stream>>>(f_den, 0.f, (long long)N);
        fill_f32_k<<<cdiv_ll((long long)N * 64, B), B, 0, stream>>>(f_out2, 0.f, (long long)N * 64);
        edge_max_k<<<cdiv_ll(Etot, B), B, 0, stream>>>(ei, E, N, f_es, f_ed, f_m, 1);
        edge_exp_k<<<cdiv_ll(Etot, B), B, 0, stream>>>(ei, E, N, f_es, f_ed, f_m, f_den, f_alp, 1);
        edge_scatter_k<<<cdiv_ll(Etot * C, B), B, 0, stream>>>(ei, E, N, f_h, f_alp, f_den, f_out2, 1, C);
        bias_act_k<ACT_NONE><<<cdiv_ll((long long)N * 64, B), B, 0, stream>>>(f_out2, b2, N, 64);

        // ---------- GRU step ----------
        cvt_f16_k<<<cdiv_ll((long long)N * 64, B), B, 0, stream>>>(f_out2, h_A, (long long)N * 64);
        cvt_f16_k<<<cdiv_ll((long long)N * 64, B), B, 0, stream>>>(f_hid,  h_B, (long long)N * 64);
        gemm_wmma_k<ACT_NONE, true><<<gemm_grid(N, 192), B, 0, stream>>>(
            h_A, h_Wih, bih, f_gi, N, 64, 192);
        gemm_wmma_k<ACT_NONE, true><<<gemm_grid(N, 192), B, 0, stream>>>(
            h_B, h_Whh, bhh, f_gh, N, 64, 192);
        gru_gate_k<<<cdiv_ll((long long)N * 64, B), B, 0, stream>>>(f_gi, f_gh, f_hid, N, 64);
    }

    // ---------- MLP head ----------
    cvt_f16_k<<<cdiv_ll((long long)N * 64, B), B, 0, stream>>>(f_hid, h_A, (long long)N * 64);
    gemm_wmma_k<ACT_RELU, true><<<gemm_grid(N, 64), B, 0, stream>>>(
        h_A, h_Wm1, bm1, f_tmp, N, 64, 64);
    mlp_out_k<<<cdiv_ll((long long)N, B), B, 0, stream>>>(f_tmp, Wm2, bm2, (float*)d_out, N);
}
